// S4Fold_33775622815805
// MI455X (gfx1250) — matrix operand order repaired and checked by reading, not compiled, not run on hardware
//
#include <hip/hip_runtime.h>

// ---------------------------------------------------------------------------
// S4 U-Net pipeline for MI455X (gfx1250, wave32).
// Convolutions: implicit-GEMM over V_WMMA_F32_16X16X32_F16, double-buffered
// LDS, weights staged with gfx1250 async global->LDS loads (ASYNCcnt).
// ---------------------------------------------------------------------------

typedef __attribute__((ext_vector_type(16))) _Float16 v16h;
typedef __attribute__((ext_vector_type(8)))  _Float16 v8h;
typedef __attribute__((ext_vector_type(8)))  float    v8f;

#define WMMA_F16(a, b, c) \
  __builtin_amdgcn_wmma_f32_16x16x32_f16(false, (a), false, (b), (short)0, (c), false, false)

__device__ __forceinline__ void wait_asynccnt0() {
#if __has_builtin(__builtin_amdgcn_s_wait_asynccnt)
  __builtin_amdgcn_s_wait_asynccnt(0);
#else
  asm volatile("s_wait_asynccnt 0x0" ::: "memory");
#endif
}

// ============================= S4 front-end ================================

__global__ void s4_in_kernel(const float* __restrict__ seqs, const float* __restrict__ Wi,
                             const float* __restrict__ bi, float* __restrict__ x) {
  int idx = blockIdx.x * blockDim.x + threadIdx.x;
  if (idx >= 512 * 4) return;
  int l = idx >> 2, d = idx & 3;
  float s = bi[d];
#pragma unroll
  for (int j = 0; j < 4; ++j) s += seqs[l * 4 + j] * Wi[d * 4 + j];
  x[idx] = s;
}

// k[d,t] = sum_n C*B*dt * exp(-exp(A_log)*dt)^t
__global__ void s4_kern_kernel(const float* __restrict__ A_log, const float* __restrict__ Bm,
                               const float* __restrict__ Cm, const float* __restrict__ dt,
                               float* __restrict__ kk) {
  int idx = blockIdx.x * blockDim.x + threadIdx.x;
  if (idx >= 4 * 512) return;
  int d = idx >> 9, t = idx & 511;
  float dtv = dt[d];
  float s = 0.0f;
  for (int n = 0; n < 64; ++n) {
    float a = -__expf(A_log[d * 64 + n]);
    float coef = Cm[d * 64 + n] * Bm[d * 64 + n] * dtv;
    s += coef * __expf(a * dtv * (float)t);
  }
  kk[idx] = s;
}

// causal conv: y[t,d] = sum_{s<=t} k[d,t-s]*u[s,d] + D[d]*u[t,d]
__global__ void s4_conv_kernel(const float* __restrict__ kk, const float* __restrict__ x,
                               const float* __restrict__ Dp, float* __restrict__ y) {
  int idx = blockIdx.x * blockDim.x + threadIdx.x;
  if (idx >= 512 * 4) return;
  int t = idx >> 2, d = idx & 3;
  const float* kd = kk + d * 512;
  float s = 0.0f;
  for (int q = 0; q <= t; ++q) s += kd[t - q] * x[q * 4 + d];
  y[idx] = s + Dp[d] * x[t * 4 + d];
}

__global__ void s4_out_kernel(const float* __restrict__ y, const float* __restrict__ Wo,
                              const float* __restrict__ bo, float* __restrict__ o) {
  int idx = blockIdx.x * blockDim.x + threadIdx.x;
  if (idx >= 512 * 4) return;
  int l = idx >> 2, d = idx & 3;
  float s = bo[d];
#pragma unroll
  for (int j = 0; j < 4; ++j) s += y[l * 4 + j] * Wo[d * 4 + j];
  o[idx] = s;
}

__global__ void ln_kernel(const float* __restrict__ x, const float* __restrict__ g,
                          const float* __restrict__ b, float* __restrict__ o) {
  int l = blockIdx.x * blockDim.x + threadIdx.x;
  if (l >= 512) return;
  float a0 = x[l * 4], a1 = x[l * 4 + 1], a2 = x[l * 4 + 2], a3 = x[l * 4 + 3];
  float m = (a0 + a1 + a2 + a3) * 0.25f;
  float d0 = a0 - m, d1 = a1 - m, d2 = a2 - m, d3 = a3 - m;
  float v = (d0 * d0 + d1 * d1 + d2 * d2 + d3 * d3) * 0.25f;
  float is = rsqrtf(v + 1e-5f);
  o[l * 4 + 0] = g[0] * d0 * is + b[0];
  o[l * 4 + 1] = g[1] * d1 * is + b[1];
  o[l * 4 + 2] = g[2] * d2 * is + b[2];
  o[l * 4 + 3] = g[3] * d3 * is + b[3];
}

// cov (torch.cov over 4 samples, /3) then x*sigmoid(x); writes channel 0 of in3
__global__ void cov_silu_kernel(const float* __restrict__ ns, float* __restrict__ out) {
  int idx = blockIdx.x * blockDim.x + threadIdx.x;
  if (idx >= 512 * 512) return;
  int i = idx >> 9, j = idx & 511;
  float mi = (ns[i * 4] + ns[i * 4 + 1] + ns[i * 4 + 2] + ns[i * 4 + 3]) * 0.25f;
  float mj = (ns[j * 4] + ns[j * 4 + 1] + ns[j * 4 + 2] + ns[j * 4 + 3]) * 0.25f;
  float s = 0.0f;
#pragma unroll
  for (int c = 0; c < 4; ++c) s += (ns[i * 4 + c] - mi) * (ns[j * 4 + c] - mj);
  s *= (1.0f / 3.0f);
  out[idx] = s * (1.0f / (1.0f + __expf(-s)));
}

// ============================ conv machinery ===============================

// fp32 OIHW weights -> f16 [Cout, Kpad] (K = Cin*9, zero-padded to 32)
__global__ void pack_w_kernel(const float* __restrict__ w, _Float16* __restrict__ wp,
                              int K, int Kpad, int n) {
  int idx = blockIdx.x * blockDim.x + threadIdx.x;
  if (idx >= n) return;
  int co = idx / Kpad, k = idx - co * Kpad;
  wp[idx] = (_Float16)(k < K ? w[co * K + k] : 0.0f);
}

// Implicit-GEMM 3x3 conv, pad=1. Block = 256 thr (8 wave32), tile 64(M)x64(N).
// Double-buffered LDS, 1 barrier/iter; weights via async global->LDS.
__launch_bounds__(256, 2)
__global__ void conv3x3_wmma_kernel(const float* __restrict__ X,
                                    const _Float16* __restrict__ Wp,
                                    const float* __restrict__ bias,
                                    float* __restrict__ Y,
                                    int Cin, int Cout, int H, int Wd,
                                    int wshift, int Kpad) {
  constexpr int LDA = 40;  // halfs per LDS row (32 + 8 pad -> 80B stride)
  __shared__ _Float16 ldsA[2][64 * LDA] __attribute__((aligned(16)));
  __shared__ _Float16 ldsB[2][64 * LDA] __attribute__((aligned(16)));

  const int HW = H * Wd;
  const int tid = threadIdx.x;
  const int wv = tid >> 5;        // wave 0..7 (wave32)
  const int lane = tid & 31;
  const int hi = lane >> 4;
  const int lr = lane & 15;
  const int mi = wv >> 1;         // M sub-tile 0..3
  const int nj0 = (wv & 1) * 2;   // first of two N sub-tiles

  const int n0 = blockIdx.x * 64;
  const int cout0 = blockIdx.y * 64;

  v8f acc0 = {};
  v8f acc1 = {};

  // A staging: 4 threads per row, 16B each (contiguous within a row)
  const int am = tid >> 2;
  const int ak = (tid & 3) * 8;
  const bool aok = (cout0 + am) < Cout;
  const _Float16* aGlob = Wp + (size_t)(cout0 + am) * (size_t)Kpad + ak;

  // B staging: adjacent lanes = adjacent pixels (coalesced X reads), 8 k each
  const int bn = tid & 63;
  const int bk = (tid >> 6) * 8;
  const int p = n0 + bn;
  const bool pok = p < HW;
  const int ypix = p >> wshift;
  const int xpix = p & (Wd - 1);

  // incremental k -> (ci, r=k%9) decode state for the 8 k-slots (advance +32)
  int ciS[8], rS[8];
#pragma unroll
  for (int i = 0; i < 8; ++i) {
    int k = bk + i;
    ciS[i] = k / 9;
    rS[i] = k - 9 * ciS[i];
  }

  const int nk = Kpad >> 5;

  auto stage = [&](int k0, int buf) {
    // ---- A tile: async global->LDS, tracked by ASYNCcnt ----
    if (aok) {
      unsigned int la = (unsigned int)(size_t)&ldsA[buf][am * LDA + ak];
      unsigned long long ga = (unsigned long long)(size_t)(aGlob + k0);
      asm volatile("global_load_async_to_lds_b128 %0, %1, off"
                   :: "v"(la), "v"(ga) : "memory");
    } else {
#pragma unroll
      for (int i = 0; i < 8; ++i) ldsA[buf][am * LDA + ak + i] = (_Float16)0.0f;
    }
    // ---- B tile: im2col gather, f32 -> f16 ----
#pragma unroll
    for (int i = 0; i < 8; ++i) {
      int ci = ciS[i];
      int r = rS[i];
      int ky = (r >= 6) ? 2 : ((r >= 3) ? 1 : 0);
      int kx = r - 3 * ky;
      float v = 0.0f;
      if (pok && ci < Cin) {
        int iy = ypix + ky - 1;
        int ix = xpix + kx - 1;
        if (iy >= 0 && iy < H && ix >= 0 && ix < Wd)
          v = X[(size_t)ci * HW + (iy << wshift) + ix];
      }
      ldsB[buf][bn * LDA + bk + i] = (_Float16)v;
      // advance decode by +32 (32 = 3*9 + 5)
      int rn = r + 5;
      int carry = (rn >= 9) ? 1 : 0;
      rS[i] = rn - (carry ? 9 : 0);
      ciS[i] = ci + 3 + carry;
    }
  };

  stage(0, 0);

  for (int it = 0; it < nk; ++it) {
    const int cb = it & 1;
    wait_asynccnt0();     // this wave's async stage into cb has landed
    __syncthreads();      // everyone's stage into cb done; prior reads of cb^1 done
    if (it + 1 < nk) stage((it + 1) << 5, cb ^ 1);

    const _Float16* As = &ldsA[cb][0];
    const _Float16* Bs = &ldsB[cb][0];

    const int mrow = mi * 16 + lr;
    v8h a0v = *(const v8h*)&As[mrow * LDA + hi * 16];
    v8h a1v = *(const v8h*)&As[mrow * LDA + hi * 16 + 8];
    v16h afrag = __builtin_shufflevector(a0v, a1v, 0, 1, 2, 3, 4, 5, 6, 7,
                                         8, 9, 10, 11, 12, 13, 14, 15);

    const int nrow0 = nj0 * 16 + lr;
    v8h b00 = *(const v8h*)&Bs[nrow0 * LDA + hi * 16];
    v8h b01 = *(const v8h*)&Bs[nrow0 * LDA + hi * 16 + 8];
    v16h bf0 = __builtin_shufflevector(b00, b01, 0, 1, 2, 3, 4, 5, 6, 7,
                                       8, 9, 10, 11, 12, 13, 14, 15);

    const int nrow1 = (nj0 + 1) * 16 + lr;
    v8h b10 = *(const v8h*)&Bs[nrow1 * LDA + hi * 16];
    v8h b11 = *(const v8h*)&Bs[nrow1 * LDA + hi * 16 + 8];
    v16h bf1 = __builtin_shufflevector(b10, b11, 0, 1, 2, 3, 4, 5, 6, 7,
                                       8, 9, 10, 11, 12, 13, 14, 15);

    acc0 = WMMA_F16(afrag, bf0, acc0);
    acc1 = WMMA_F16(afrag, bf1, acc1);
  }

  // ---- epilogue: D layout = VGPR r -> M = r + 8*hi, N = lane&15 ----
  const int cobase = cout0 + mi * 16 + hi * 8;
  const int p0 = n0 + nj0 * 16 + lr;
  const int p1 = n0 + (nj0 + 1) * 16 + lr;
#pragma unroll
  for (int r = 0; r < 8; ++r) {
    int c = cobase + r;
    if (c < Cout) {
      float bb = bias ? bias[c] : 0.0f;
      if (p0 < HW) Y[(size_t)c * HW + p0] = acc0[r] + bb;
      if (p1 < HW) Y[(size_t)c * HW + p1] = acc1[r] + bb;
    }
  }
}

// =========================== BN / pool / resize ============================

__global__ void bn_stats_kernel(const float* __restrict__ X, float* __restrict__ mean,
                                float* __restrict__ istd, int HW) {
  __shared__ float sh0[256];
  __shared__ float sh1[256];
  int c = blockIdx.x;
  const float* p = X + (size_t)c * HW;
  float s = 0.0f, s2 = 0.0f;
  for (int i = threadIdx.x; i < HW; i += 256) {
    float v = p[i];
    s += v;
    s2 += v * v;
  }
  sh0[threadIdx.x] = s;
  sh1[threadIdx.x] = s2;
  __syncthreads();
  for (int off = 128; off > 0; off >>= 1) {
    if (threadIdx.x < off) {
      sh0[threadIdx.x] += sh0[threadIdx.x + off];
      sh1[threadIdx.x] += sh1[threadIdx.x + off];
    }
    __syncthreads();
  }
  if (threadIdx.x == 0) {
    float m = sh0[0] / (float)HW;
    float v = sh1[0] / (float)HW - m * m;
    mean[c] = m;
    istd[c] = rsqrtf(v + 1e-5f);
  }
}

__global__ void bn_apply_relu_kernel(float* __restrict__ Y, const float* __restrict__ g,
                                     const float* __restrict__ be, const float* __restrict__ mean,
                                     const float* __restrict__ istd, int hwshift, int n) {
  int idx = blockIdx.x * blockDim.x + threadIdx.x;
  if (idx >= n) return;
  int c = idx >> hwshift;
  float v = (Y[idx] - mean[c]) * istd[c] * g[c] + be[c];
  Y[idx] = v > 0.0f ? v : 0.0f;
}

__global__ void maxpool_kernel(const float* __restrict__ X, float* __restrict__ Y,
                               int H, int Wd, int n) {
  int idx = blockIdx.x * blockDim.x + threadIdx.x;
  if (idx >= n) return;
  int Wo = Wd >> 1, Ho = H >> 1;
  int c = idx / (Ho * Wo);
  int rem = idx - c * Ho * Wo;
  int y = rem / Wo, x = rem - y * Wo;
  const float* p = X + (size_t)c * H * Wd + (size_t)(2 * y) * Wd + 2 * x;
  Y[idx] = fmaxf(fmaxf(p[0], p[1]), fmaxf(p[Wd], p[Wd + 1]));
}

__global__ void upsample_kernel(const float* __restrict__ X, float* __restrict__ Y,
                                int Ho, int Wo, int n) {  // output dims, nearest 2x
  int idx = blockIdx.x * blockDim.x + threadIdx.x;
  if (idx >= n) return;
  int c = idx / (Ho * Wo);
  int rem = idx - c * Ho * Wo;
  int y = rem / Wo, x = rem - y * Wo;
  int Hi = Ho >> 1, Wi = Wo >> 1;
  Y[idx] = X[(size_t)c * Hi * Wi + (size_t)(y >> 1) * Wi + (x >> 1)];
}

// 1x1 conv, 32 -> 1
__global__ void readout_kernel(const float* __restrict__ X, const float* __restrict__ w,
                               const float* __restrict__ b, float* __restrict__ Y, int HW) {
  int p = blockIdx.x * blockDim.x + threadIdx.x;
  if (p >= HW) return;
  float s = b[0];
#pragma unroll
  for (int c = 0; c < 32; ++c) s += w[c] * X[(size_t)c * HW + p];
  Y[p] = s;
}

// out[i,j] = y[j,i] * y[i,j]
__global__ void final_kernel(const float* __restrict__ y, float* __restrict__ out) {
  int idx = blockIdx.x * blockDim.x + threadIdx.x;
  if (idx >= 512 * 512) return;
  int i = idx >> 9, j = idx & 511;
  out[idx] = y[j * 512 + i] * y[idx];
}

// ================================ host side ================================

extern "C" void kernel_launch(void* const* d_in, const int* in_sizes, int n_in,
                              void* d_out, int out_size, void* d_ws, size_t ws_size,
                              hipStream_t stream) {
  (void)in_sizes; (void)n_in; (void)out_size; (void)ws_size;
  const float* const* in = (const float* const*)d_in;

  // setup_inputs() flattening order:
  const float* seqs = in[0];
  const float* cosm = in[1];
  const float* fmap = in[2];
  const float* Wi = in[3];  const float* bi = in[4];
  const float* Alog = in[5]; const float* Bm = in[6]; const float* Cm = in[7];
  const float* Dp = in[8];  const float* dt = in[9];
  const float* Wo = in[10]; const float* bo = in[11];
  const float* lng = in[12]; const float* lnb = in[13];
  auto encP = [&](int blk, int j) { return in[14 + blk * 8 + j]; };
  auto decP = [&](int blk, int j) { return in[54 + blk * 12 + j]; };
  const float* roW = in[102];
  const float* roB = in[103];

  // deterministic bump allocator over d_ws (~620 MB total footprint)
  size_t off = 0;
  char* base = (char*)d_ws;
  auto alloc = [&](size_t bytes) -> void* {
    off = (off + 255) & ~(size_t)255;
    void* pp = base + off;
    off += bytes;
    return pp;
  };

  float* bnMean = (float*)alloc(512 * sizeof(float));
  float* bnIstd = (float*)alloc(512 * sizeof(float));

  auto conv_bn_relu = [&](const float* Xin, const float* w, const float* cb,
                          const float* g, const float* be, float* Yout,
                          int Cin, int Cout, int Hh, int Ww) {
    int K = Cin * 9, Kpad = (K + 31) & ~31;
    _Float16* wp = (_Float16*)alloc((size_t)Cout * Kpad * sizeof(_Float16));
    int npack = Cout * Kpad;
    pack_w_kernel<<<(npack + 255) / 256, 256, 0, stream>>>(w, wp, K, Kpad, npack);
    int wsh = 31 - __builtin_clz((unsigned)Ww);
    int HWp = Hh * Ww;
    dim3 grid((HWp + 63) / 64, (Cout + 63) / 64);
    conv3x3_wmma_kernel<<<grid, 256, 0, stream>>>(Xin, wp, cb, Yout, Cin, Cout, Hh, Ww, wsh, Kpad);
    bn_stats_kernel<<<Cout, 256, 0, stream>>>(Yout, bnMean, bnIstd, HWp);
    int hwsh = 31 - __builtin_clz((unsigned)HWp);
    int n = Cout * HWp;
    bn_apply_relu_kernel<<<(n + 255) / 256, 256, 0, stream>>>(Yout, g, be, bnMean, bnIstd, hwsh, n);
  };

  // ---------------- S4 + cov front end ----------------
  float* s4x = (float*)alloc(512 * 4 * 4);
  float* s4k = (float*)alloc(4 * 512 * 4);
  float* s4y = (float*)alloc(512 * 4 * 4);
  float* s4o = (float*)alloc(512 * 4 * 4);
  float* nsb = (float*)alloc(512 * 4 * 4);
  float* in3 = (float*)alloc((size_t)3 * 512 * 512 * 4);

  s4_in_kernel<<<8, 256, 0, stream>>>(seqs, Wi, bi, s4x);
  s4_kern_kernel<<<8, 256, 0, stream>>>(Alog, Bm, Cm, dt, s4k);
  s4_conv_kernel<<<8, 256, 0, stream>>>(s4k, s4x, Dp, s4y);
  s4_out_kernel<<<8, 256, 0, stream>>>(s4y, Wo, bo, s4o);
  ln_kernel<<<2, 256, 0, stream>>>(s4o, lng, lnb, nsb);
  cov_silu_kernel<<<(512 * 512 + 255) / 256, 256, 0, stream>>>(nsb, in3);
  hipMemcpyAsync(in3 + 512 * 512, cosm, (size_t)512 * 512 * 4, hipMemcpyDeviceToDevice, stream);
  hipMemcpyAsync(in3 + 2 * 512 * 512, fmap, (size_t)512 * 512 * 4, hipMemcpyDeviceToDevice, stream);

  // ---------------- encoder ----------------
  int Cs[6] = {3, 32, 64, 128, 256, 512};
  int Hs[5] = {512, 256, 128, 64, 32};
  float* skip[5];
  const float* cur = in3;
  for (int s = 0; s < 5; ++s) {
    int Hh = Hs[s], Ci = Cs[s], Co = Cs[s + 1];
    if (s > 0) {
      float* pooled = (float*)alloc((size_t)Ci * Hh * Hh * 4);
      int n = Ci * Hh * Hh;
      maxpool_kernel<<<(n + 255) / 256, 256, 0, stream>>>(cur, pooled, Hs[s - 1], Hs[s - 1], n);
      cur = pooled;
    }
    float* a = (float*)alloc((size_t)Co * Hh * Hh * 4);
    float* b2 = (float*)alloc((size_t)Co * Hh * Hh * 4);
    conv_bn_relu(cur, encP(s, 0), encP(s, 1), encP(s, 2), encP(s, 3), a, Ci, Co, Hh, Hh);
    conv_bn_relu(a, encP(s, 4), encP(s, 5), encP(s, 6), encP(s, 7), b2, Co, Co, Hh, Hh);
    skip[s] = b2;
    cur = b2;
  }

  // ---------------- decoder ----------------
  int Cd[5] = {512, 256, 128, 64, 32};
  int Hd = 32;
  for (int s = 0; s < 4; ++s) {
    int Ci = Cd[s], Co = Cd[s + 1];
    int Ho = Hd * 2;
    float* up = (float*)alloc((size_t)Ci * Ho * Ho * 4);
    int n = Ci * Ho * Ho;
    upsample_kernel<<<(n + 255) / 256, 256, 0, stream>>>(cur, up, Ho, Ho, n);
    float* upc = (float*)alloc((size_t)Co * Ho * Ho * 4);
    conv_bn_relu(up, decP(s, 0), decP(s, 1), decP(s, 2), decP(s, 3), upc, Ci, Co, Ho, Ho);
    float* cat = (float*)alloc((size_t)2 * Co * Ho * Ho * 4);
    hipMemcpyAsync(cat, upc, (size_t)Co * Ho * Ho * 4, hipMemcpyDeviceToDevice, stream);
    hipMemcpyAsync(cat + (size_t)Co * Ho * Ho, skip[3 - s], (size_t)Co * Ho * Ho * 4,
                   hipMemcpyDeviceToDevice, stream);
    float* c1 = (float*)alloc((size_t)Co * Ho * Ho * 4);
    float* c2 = (float*)alloc((size_t)Co * Ho * Ho * 4);
    conv_bn_relu(cat, decP(s, 4), decP(s, 5), decP(s, 6), decP(s, 7), c1, 2 * Co, Co, Ho, Ho);
    conv_bn_relu(c1, decP(s, 8), decP(s, 9), decP(s, 10), decP(s, 11), c2, Co, Co, Ho, Ho);
    cur = c2;
    Hd = Ho;
  }

  // ---------------- readout + transpose-multiply ----------------
  float* yro = (float*)alloc((size_t)512 * 512 * 4);
  readout_kernel<<<(512 * 512 + 255) / 256, 256, 0, stream>>>(cur, roW, roB, yro, 512 * 512);
  final_kernel<<<(512 * 512 + 255) / 256, 256, 0, stream>>>(yro, (float*)d_out);
}
